// Decoder_65060164600142
// MI455X (gfx1250) — compile-verified, matrix operand order checked by reading, not run on hardware
//
#include <hip/hip_runtime.h>
#include <hip/hip_bf16.h>

// ---------------------------------------------------------------------------
// Pairwise squared distance + sigmoid:
//   out[i,j] = sigmoid(alpha - 0.5*(||x_i||^2 + ||y_j||^2 - 2 x_i.y_j))
// X:[8192,512] f32, Y:[8192,512] f32  ->  out:[8192,8192] f32
// GEMM via v_wmma_f32_16x16x32_bf16; tile fill via GLOBAL_LOAD_ASYNC_TO_LDS
// (no VGPR staging -> no spills), norms in fp32, NT stores for the 256MB out.
// ---------------------------------------------------------------------------

typedef __attribute__((ext_vector_type(16))) __bf16          v16bf;
typedef __attribute__((ext_vector_type(8)))  float           v8f;
typedef __attribute__((ext_vector_type(8)))  unsigned short  v8u16;

#define N1  8192
#define N3  8192
#define DIM 512

#define BM 128          // block tile M
#define BN 128          // block tile N
#define KC 64           // K chunk held in LDS (bf16 elements)
#define LDS_STRIDE 72   // padded row stride in bf16 elems (144B, 16B aligned)

#define NCHUNK (DIM / KC)

struct Frag32 { v8u16 lo; v8u16 hi; };   // 32 bytes == v16bf

#if __has_builtin(__builtin_amdgcn_global_load_async_to_lds_b128)
#define USE_ASYNC_LDS 1
#endif

#ifdef USE_ASYNC_LDS
// Builtin signature (from clang diagnostic): param 0 is
// 'int __attribute__((vector_size(16))) __device__ *' i.e. AS1 int4*.
typedef int vec4i __attribute__((vector_size(16)));
typedef __attribute__((address_space(1))) vec4i as1_vec4i;
typedef __attribute__((address_space(3))) vec4i as3_vec4i;

__device__ __forceinline__ void async_copy_b128(const unsigned short* g,
                                                unsigned short* l)
{
    as1_vec4i* gp = (as1_vec4i*)(unsigned long long)(uintptr_t)g;
    // low 32 bits of a generic LDS pointer == LDS byte offset (aperture map)
    as3_vec4i* lp = (as3_vec4i*)(unsigned int)(uintptr_t)l;
    __builtin_amdgcn_global_load_async_to_lds_b128(gp, lp, 0, 0);
}
#endif

__device__ __forceinline__ void wait_async_zero()
{
#if __has_builtin(__builtin_amdgcn_s_wait_asynccnt)
    __builtin_amdgcn_s_wait_asynccnt(0);
#else
    asm volatile("s_wait_asynccnt 0x0" ::: "memory");
#endif
}

// ---------------------------------------------------------------------------
// prep: per-row fp32 squared norm + fp32 -> bf16 conversion into workspace
// one block (128 threads) per row; rows [0,N1) = X, rows [N1,N1+N3) = Y
// ---------------------------------------------------------------------------
__global__ __launch_bounds__(128)
void prep_kernel(const float* __restrict__ X, const float* __restrict__ Y,
                 unsigned short* __restrict__ Xb, unsigned short* __restrict__ Yb,
                 float* __restrict__ xsq, float* __restrict__ ysq)
{
    const int row = blockIdx.x;
    const float* src; unsigned short* dst; float* nrm; int r;
    if (row < N1) { src = X; dst = Xb; nrm = xsq; r = row; }
    else          { src = Y; dst = Yb; nrm = ysq; r = row - N1; }

    const int t = threadIdx.x;                       // 128 threads * float4 = 512
    const float4 v = ((const float4*)(src + (size_t)r * DIM))[t];
    float p = v.x*v.x + v.y*v.y + v.z*v.z + v.w*v.w;

    ushort4 b;
    b.x = __builtin_bit_cast(unsigned short, (__bf16)v.x);
    b.y = __builtin_bit_cast(unsigned short, (__bf16)v.y);
    b.z = __builtin_bit_cast(unsigned short, (__bf16)v.z);
    b.w = __builtin_bit_cast(unsigned short, (__bf16)v.w);
    ((ushort4*)(dst + (size_t)r * DIM))[t] = b;

    #pragma unroll
    for (int off = 16; off > 0; off >>= 1)
        p += __shfl_down(p, off, 32);
    __shared__ float red[4];
    if ((t & 31) == 0) red[t >> 5] = p;
    __syncthreads();
    if (t == 0) nrm[r] = red[0] + red[1] + red[2] + red[3];
}

// ---------------------------------------------------------------------------
// fused distance GEMM: 128x128 tile / block, 8 waves in 4(M) x 2(N) grid,
// each wave -> 32x64 sub-tile = 2x4 wmma 16x16 accumulators.
// ---------------------------------------------------------------------------
__global__ __launch_bounds__(256)
__attribute__((amdgpu_waves_per_eu(2)))
void dist_gemm_kernel(const unsigned short* __restrict__ Xb,
                      const unsigned short* __restrict__ Yb,
                      const float* __restrict__ xsq,
                      const float* __restrict__ ysq,
                      const float* __restrict__ alpha_p,
                      float* __restrict__ out)
{
    __shared__ unsigned short lA[2][BM * LDS_STRIDE];
    __shared__ unsigned short lB[2][BN * LDS_STRIDE];

    const int tid   = threadIdx.x;
    const int bm    = blockIdx.x * BM;
    const int bn    = blockIdx.y * BN;
    const int lane  = tid & 31;
    const int wave  = tid >> 5;
    const int wm    = (wave & 3) * 32;   // wave M offset within tile
    const int wn    = (wave >> 2) * 64;  // wave N offset within tile
    const int lhalf = lane >> 4;         // 0: lanes 0-15, 1: lanes 16-31
    const int l15   = lane & 15;

#ifdef USE_ASYNC_LDS
    // global -> LDS directly, 4 x 16B segments per thread per matrix
    auto issue_chunk = [&](int k0, int s) {
        #pragma unroll
        for (int i = 0; i < 4; ++i) {
            const int seg = tid + i * 256;
            const int row = seg >> 3;
            const int col = (seg & 7) * 8;
            async_copy_b128(Xb + (size_t)(bm + row) * DIM + k0 + col,
                            &lA[s][row * LDS_STRIDE + col]);
            async_copy_b128(Yb + (size_t)(bn + row) * DIM + k0 + col,
                            &lB[s][row * LDS_STRIDE + col]);
        }
    };
#else
    uint4 ra[4], rb[4];
    auto load_chunk = [&](int k0) {
        #pragma unroll
        for (int i = 0; i < 4; ++i) {
            const int seg = tid + i * 256;
            const int row = seg >> 3;
            const int col = (seg & 7) * 8;
            ra[i] = *(const uint4*)(Xb + (size_t)(bm + row) * DIM + k0 + col);
            rb[i] = *(const uint4*)(Yb + (size_t)(bn + row) * DIM + k0 + col);
        }
    };
    auto store_chunk = [&](int s) {
        #pragma unroll
        for (int i = 0; i < 4; ++i) {
            const int seg = tid + i * 256;
            const int row = seg >> 3;
            const int col = (seg & 7) * 8;
            *(uint4*)&lA[s][row * LDS_STRIDE + col] = ra[i];
            *(uint4*)&lB[s][row * LDS_STRIDE + col] = rb[i];
        }
    };
#endif

    v8f acc[2][4];
    #pragma unroll
    for (int i = 0; i < 2; ++i)
        #pragma unroll
        for (int j = 0; j < 4; ++j)
            acc[i][j] = {};

    // Fragment gather per documented CDNA5 16-bit A/B layout:
    // lane l (l15 = row), elems 0..7  -> K = kk + lhalf*8 + 0..7
    //                     elems 8..15 -> K = kk + 16 + lhalf*8 + 0..7
    auto compute_stage = [&](int s) {
        #pragma unroll
        for (int kk = 0; kk < KC; kk += 32) {
            v16bf a[2], b[4];
            #pragma unroll
            for (int mt = 0; mt < 2; ++mt) {
                const unsigned short* p =
                    &lA[s][(wm + mt * 16 + l15) * LDS_STRIDE + kk + lhalf * 8];
                Frag32 f; f.lo = *(const v8u16*)p; f.hi = *(const v8u16*)(p + 16);
                a[mt] = __builtin_bit_cast(v16bf, f);
            }
            #pragma unroll
            for (int nt = 0; nt < 4; ++nt) {
                const unsigned short* p =
                    &lB[s][(wn + nt * 16 + l15) * LDS_STRIDE + kk + lhalf * 8];
                Frag32 f; f.lo = *(const v8u16*)p; f.hi = *(const v8u16*)(p + 16);
                b[nt] = __builtin_bit_cast(v16bf, f);
            }
            #pragma unroll
            for (int mt = 0; mt < 2; ++mt)
                #pragma unroll
                for (int nt = 0; nt < 4; ++nt)
                    acc[mt][nt] = __builtin_amdgcn_wmma_f32_16x16x32_bf16(
                        false, a[mt], false, b[nt],
                        (short)0, acc[mt][nt], false, false);
        }
    };

    // Double-buffered K loop: 512 / 64 = 8 chunks
#ifdef USE_ASYNC_LDS
    issue_chunk(0, 0);
    wait_async_zero();
    __syncthreads();
    #pragma unroll 1
    for (int c = 0; c < NCHUNK; ++c) {
        if (c + 1 < NCHUNK) issue_chunk((c + 1) * KC, (c + 1) & 1);
        compute_stage(c & 1);
        if (c + 1 < NCHUNK) wait_async_zero();
        __syncthreads();
    }
#else
    load_chunk(0);
    store_chunk(0);
    __syncthreads();
    #pragma unroll 1
    for (int c = 0; c < NCHUNK; ++c) {
        if (c + 1 < NCHUNK) load_chunk((c + 1) * KC);
        compute_stage(c & 1);
        if (c + 1 < NCHUNK) store_chunk((c + 1) & 1);
        __syncthreads();
    }
#endif

    // Epilogue: alpha - 0.5*xsq - 0.5*ysq + (x.y), then sigmoid, NT store.
    // C/D layout: VGPR v, lanes 0-15: (M=v, N=lane); lanes 16-31: (M=v+8, N=lane-16)
    const float alpha = *alpha_p;
    float ysv[4];
    #pragma unroll
    for (int nt = 0; nt < 4; ++nt)
        ysv[nt] = ysq[bn + wn + nt * 16 + l15];

    #pragma unroll
    for (int mt = 0; mt < 2; ++mt) {
        #pragma unroll
        for (int v = 0; v < 8; ++v) {
            const int m = bm + wm + mt * 16 + v + lhalf * 8;
            const float base = alpha - 0.5f * xsq[m];
            const size_t rowoff = (size_t)m * N3;
            #pragma unroll
            for (int nt = 0; nt < 4; ++nt) {
                const int n = bn + wn + nt * 16 + l15;
                const float t = base - 0.5f * ysv[nt] + acc[mt][nt][v];
                const float r = 1.0f / (1.0f + __expf(-t));
                __builtin_nontemporal_store(r, &out[rowoff + n]);
            }
        }
    }
}

// ---------------------------------------------------------------------------
extern "C" void kernel_launch(void* const* d_in, const int* in_sizes, int n_in,
                              void* d_out, int out_size, void* d_ws, size_t ws_size,
                              hipStream_t stream)
{
    const float* X     = (const float*)d_in[0];
    const float* Y     = (const float*)d_in[1];
    const float* alpha = (const float*)d_in[2];
    float* out         = (float*)d_out;

    // workspace layout: Xb (bf16) | Yb (bf16) | xsq | ysq  (~16.8 MB)
    unsigned short* Xb = (unsigned short*)d_ws;
    unsigned short* Yb = Xb + (size_t)N1 * DIM;
    float* xsq = (float*)(Yb + (size_t)N3 * DIM);
    float* ysq = xsq + N1;

    prep_kernel<<<N1 + N3, 128, 0, stream>>>(X, Y, Xb, Yb, xsq, ysq);

    dim3 grid(N1 / BM, N3 / BN);
    dist_gemm_kernel<<<grid, 256, 0, stream>>>(Xb, Yb, xsq, ysq, alpha, out);
}
